// Wrap_Model_18305150616045
// MI455X (gfx1250) — matrix operand order repaired and checked by reading, not compiled, not run on hardware
//
#include <hip/hip_runtime.h>

// ---------------------------------------------------------------------------
// 2-NN (squared L2) on MI455X / gfx1250.
//   d2[b,n] = ||t_b||^2 + ||x_n||^2 - 2 t_b . x_n ;  out[b] = sum of 2 smallest.
//
// Cross term: 1024 x 50000 x 512 GEMM (52.4 GFLOP) -> compute-bound
// (train = 102 MB fits in 192 MB L2; one HBM pass ~4.4 us at 23.3 TB/s).
// f32 = bf16_hi + bf16_lo split; cross ~= hi*hi + hi*lo + lo*hi via
// v_wmma_f32_16x16x32_bf16 (3 instr / K=32 vs 8 for native f32 WMMA 16x16x4),
// f32 accumulate, ~16 mantissa bits, dropped lo*lo ~ 2^-16 relative.
//
// v2: split precomputed into bf16 hi/lo planes -> pure load->wmma inner loop.
// v3: inner loop restructured for the matrix pipe:
//   - all fragments of a k-step loaded before any WMMA (clause-batched loads,
//     partial loadcnt waits instead of full drains),
//   - WMMAs issued in 3 rounds of 4 sub-tiles so each accumulator is touched
//     only every 4th WMMA (no TRANS32 dependency stalls),
//   - k-loop fully unrolled: load addresses become immediate offsets on
//     loop-invariant base pointers (no per-iteration v_add_nc_u64).
// ---------------------------------------------------------------------------

typedef __bf16 v16bf __attribute__((ext_vector_type(16)));
typedef __bf16 v8bf  __attribute__((ext_vector_type(8)));
typedef __bf16 v4bf  __attribute__((ext_vector_type(4)));
typedef float  v8f   __attribute__((ext_vector_type(8)));

#define KDIM  512
#define BM    64      // test rows per block (4 WMMA sub-tiles of 16)
#define BN    128     // train cols per block (8 waves x 16 cols)
#define WAVES 8
#define BIGF  3.0e38f

// Merge sorted pair (b1<=b2) into sorted pair (a1<=a2): keep two smallest.
__device__ __forceinline__ void top2_merge(float& a1, float& a2, float b1, float b2) {
    float lo = fminf(a1, b1);
    float hi = fmaxf(a1, b1);
    a2 = fminf(fminf(a2, b2), hi);
    a1 = lo;
}

// Split 8 contiguous f32 into bf16 hi/lo, writing fragment elements [e0..e0+7].
__device__ __forceinline__ void split8(const float* __restrict__ p,
                                       v16bf& hi, v16bf& lo, int e0) {
#pragma unroll
    for (int i = 0; i < 8; ++i) {
        float x  = p[i];
        __bf16 h = (__bf16)x;
        hi[e0 + i] = h;
        lo[e0 + i] = (__bf16)(x - (float)h);
    }
}

// ---------------------------------------------------------------------------
// Kernel 0: pre-split rows into bf16 hi/lo planes; optionally row sum-of-sq.
// ---------------------------------------------------------------------------
__global__ __launch_bounds__(256) void split_rows(
    const float* __restrict__ src, __bf16* __restrict__ hip_,
    __bf16* __restrict__ lop_, float* __restrict__ sumsq, int nrows)
{
    const int wave = threadIdx.x >> 5;
    const int lane = threadIdx.x & 31;
    const int row  = blockIdx.x * WAVES + wave;
    if (row >= nrows) return;
    const float4* p  = (const float4*)(src + (size_t)row * KDIM);
    v4bf* hr = (v4bf*)(hip_ + (size_t)row * KDIM);
    v4bf* lr = (v4bf*)(lop_ + (size_t)row * KDIM);
    float s = 0.f;
#pragma unroll
    for (int i = 0; i < KDIM / 128; ++i) {
        float4 v = p[lane + i * 32];
        s += v.x * v.x + v.y * v.y + v.z * v.z + v.w * v.w;
        v4bf h, l;
        h[0] = (__bf16)v.x; l[0] = (__bf16)(v.x - (float)h[0]);
        h[1] = (__bf16)v.y; l[1] = (__bf16)(v.y - (float)h[1]);
        h[2] = (__bf16)v.z; l[2] = (__bf16)(v.z - (float)h[2]);
        h[3] = (__bf16)v.w; l[3] = (__bf16)(v.w - (float)h[3]);
        hr[lane + i * 32] = h;
        lr[lane + i * 32] = l;
    }
#pragma unroll
    for (int m = 16; m >= 1; m >>= 1) s += __shfl_xor(s, m, 32);
    if (lane == 0 && sumsq) sumsq[row] = s;
}

// Plain per-row sum of squares (fallback path).
__global__ __launch_bounds__(256) void row_sumsq(const float* __restrict__ x,
                                                 float* __restrict__ out, int nrows) {
    const int wave = threadIdx.x >> 5;
    const int lane = threadIdx.x & 31;
    const int row  = blockIdx.x * WAVES + wave;
    if (row >= nrows) return;
    const float4* p = (const float4*)(x + (size_t)row * KDIM);
    float s = 0.f;
#pragma unroll
    for (int i = 0; i < KDIM / 128; ++i) {
        float4 v = p[lane + i * 32];
        s += v.x * v.x + v.y * v.y + v.z * v.z + v.w * v.w;
    }
#pragma unroll
    for (int m = 16; m >= 1; m >>= 1) s += __shfl_xor(s, m, 32);
    if (lane == 0) out[row] = s;
}

__device__ __forceinline__ v16bf cat8(v8bf a, v8bf b) {
    return __builtin_shufflevector(a, b, 0, 1, 2, 3, 4, 5, 6, 7,
                                   8, 9, 10, 11, 12, 13, 14, 15);
}

// ---------------------------------------------------------------------------
// Kernel 2: WMMA cross tile + streaming per-row top-2 over this N-chunk.
// Block: 256 threads (8 waves). Tile: BM x BN. Wave w owns cols n0w..n0w+15
// and 4 row sub-tiles; B-fragment reused across the 4 A sub-tiles.
// ---------------------------------------------------------------------------
template <bool PRESPLIT>
__global__ __launch_bounds__(256) void knn_cross_top2(
    const float*  __restrict__ testF,  const float*  __restrict__ trainF,
    const __bf16* __restrict__ testHi, const __bf16* __restrict__ testLo,
    const __bf16* __restrict__ trainHi, const __bf16* __restrict__ trainLo,
    const float* __restrict__ bsq, float2* __restrict__ partial,
    int B, int N, int nChunks)
{
    __shared__ float2 sTop[BM][WAVES];

    const int tid  = threadIdx.x;
    const int wave = tid >> 5;
    const int lane = tid & 31;
    const int l16  = lane & 15;   // column (B) / row (A) within 16-group
    const int kh   = lane >> 4;   // K-half selector per WMMA fragment layout

    const int chunk = blockIdx.x;
    const int m0    = blockIdx.y * BM;
    const int n0w   = chunk * BN + wave * 16;

    for (int i = tid; i < BM * WAVES; i += blockDim.x)
        ((float2*)sTop)[i] = make_float2(BIGF, BIGF);
    __syncthreads();

    if (n0w < N) {                      // wave-uniform: EXEC stays all-ones
        const int  ncol  = n0w + l16;
        const bool colOK = (ncol < N);
        const int  ncls  = colOK ? ncol : (N - 1);          // data-level clamp
        const float bsq_lane = bsq[ncls];

        // Clamped row index per sub-tile (data-level, EXEC untouched).
        int mrow[4];
#pragma unroll
        for (int t = 0; t < 4; ++t) {
            int r = m0 + t * 16 + l16;
            mrow[t] = (r < B) ? r : (B - 1);
        }

        v8f acc[4] = {};

        if constexpr (PRESPLIT) {
            // Loop-invariant base pointers; k becomes an immediate offset.
            const __bf16* bh = trainHi + (size_t)ncls * KDIM + kh * 8;
            const __bf16* bl = trainLo + (size_t)ncls * KDIM + kh * 8;
            const __bf16* ahp[4];
            const __bf16* alp[4];
#pragma unroll
            for (int t = 0; t < 4; ++t) {
                ahp[t] = testHi + (size_t)mrow[t] * KDIM + kh * 8;
                alp[t] = testLo + (size_t)mrow[t] * KDIM + kh * 8;
            }
#pragma unroll
            for (int k = 0; k < KDIM; k += 32) {
                // ---- load all fragments of this k-step (clause-batched) ----
                v16bf bhi = cat8(*(const v8bf*)(bh + k), *(const v8bf*)(bh + k + 16));
                v16bf blo = cat8(*(const v8bf*)(bl + k), *(const v8bf*)(bl + k + 16));
                v16bf ahi[4], alo[4];
#pragma unroll
                for (int t = 0; t < 4; ++t) {
                    ahi[t] = cat8(*(const v8bf*)(ahp[t] + k),
                                  *(const v8bf*)(ahp[t] + k + 16));
                    alo[t] = cat8(*(const v8bf*)(alp[t] + k),
                                  *(const v8bf*)(alp[t] + k + 16));
                }
                // ---- 3 rounds x 4 sub-tiles: accumulators rotate, no
                //      back-to-back same-D WMMA dependency stalls ----
#pragma unroll
                for (int t = 0; t < 4; ++t)
                    acc[t] = __builtin_amdgcn_wmma_f32_16x16x32_bf16(
                        false, ahi[t], false, bhi, (short)0, acc[t], false, false);
#pragma unroll
                for (int t = 0; t < 4; ++t)
                    acc[t] = __builtin_amdgcn_wmma_f32_16x16x32_bf16(
                        false, ahi[t], false, blo, (short)0, acc[t], false, false);
#pragma unroll
                for (int t = 0; t < 4; ++t)
                    acc[t] = __builtin_amdgcn_wmma_f32_16x16x32_bf16(
                        false, alo[t], false, bhi, (short)0, acc[t], false, false);
            }
        } else {
            const float* trow = trainF + (size_t)ncls * KDIM;
            for (int k = 0; k < KDIM; k += 32) {
                v16bf bhi, blo;
                split8(trow + k + kh * 8,      bhi, blo, 0);
                split8(trow + k + 16 + kh * 8, bhi, blo, 8);
                v16bf ahi[4], alo[4];
#pragma unroll
                for (int t = 0; t < 4; ++t) {
                    const float* arow = testF + (size_t)mrow[t] * KDIM;
                    split8(arow + k + kh * 8,      ahi[t], alo[t], 0);
                    split8(arow + k + 16 + kh * 8, ahi[t], alo[t], 8);
                }
#pragma unroll
                for (int t = 0; t < 4; ++t)
                    acc[t] = __builtin_amdgcn_wmma_f32_16x16x32_bf16(
                        false, ahi[t], false, bhi, (short)0, acc[t], false, false);
#pragma unroll
                for (int t = 0; t < 4; ++t)
                    acc[t] = __builtin_amdgcn_wmma_f32_16x16x32_bf16(
                        false, ahi[t], false, blo, (short)0, acc[t], false, false);
#pragma unroll
                for (int t = 0; t < 4; ++t)
                    acc[t] = __builtin_amdgcn_wmma_f32_16x16x32_bf16(
                        false, alo[t], false, bhi, (short)0, acc[t], false, false);
            }
        }

        // Epilogue: s = b_sq[n] - 2*cross; per-row top-2 over 16 cols.
        // C/D layout: element r of lane L = row (r + 8*(L/16)), col L%16.
#pragma unroll
        for (int t = 0; t < 4; ++t) {
#pragma unroll
            for (int r = 0; r < 8; ++r) {
                float s1 = colOK ? (bsq_lane - 2.0f * acc[t][r]) : BIGF;
                float s2 = BIGF;
#pragma unroll
                for (int m = 1; m < 16; m <<= 1) {   // butterfly in 16-lane half
                    float o1 = __shfl_xor(s1, m, 32);
                    float o2 = __shfl_xor(s2, m, 32);
                    top2_merge(s1, s2, o1, o2);
                }
                if (l16 == 0)
                    sTop[t * 16 + r + 8 * kh][wave] = make_float2(s1, s2);
            }
        }
    }
    __syncthreads();

    if (tid < BM && (m0 + tid) < B) {
        float s1 = BIGF, s2 = BIGF;
#pragma unroll
        for (int w = 0; w < WAVES; ++w) {
            float2 p = sTop[tid][w];
            top2_merge(s1, s2, p.x, p.y);
        }
        partial[(size_t)(m0 + tid) * nChunks + chunk] = make_float2(s1, s2);
    }
}

// ---------------------------------------------------------------------------
// Kernel 3: wave per test row — merge chunk partials, add 2*||t||^2.
// ---------------------------------------------------------------------------
__global__ __launch_bounds__(256) void knn_finalize(
    const float* __restrict__ test, const float2* __restrict__ partial,
    float* __restrict__ out, int B, int nChunks)
{
    const int wave = threadIdx.x >> 5;
    const int lane = threadIdx.x & 31;
    const int row  = blockIdx.x * WAVES + wave;
    if (row >= B) return;

    const float4* p = (const float4*)(test + (size_t)row * KDIM);
    float ts = 0.f;
#pragma unroll
    for (int i = 0; i < KDIM / 128; ++i) {
        float4 v = p[lane + i * 32];
        ts += v.x * v.x + v.y * v.y + v.z * v.z + v.w * v.w;
    }
#pragma unroll
    for (int m = 16; m >= 1; m >>= 1) ts += __shfl_xor(ts, m, 32);

    float s1 = BIGF, s2 = BIGF;
    const float2* pr = partial + (size_t)row * nChunks;
    for (int c = lane; c < nChunks; c += 32) {
        float2 q = pr[c];
        top2_merge(s1, s2, q.x, q.y);
    }
#pragma unroll
    for (int m = 16; m >= 1; m >>= 1) {
        float o1 = __shfl_xor(s1, m, 32);
        float o2 = __shfl_xor(s2, m, 32);
        top2_merge(s1, s2, o1, o2);
    }
    if (lane == 0) out[row] = 2.0f * ts + s1 + s2;
}

// ---------------------------------------------------------------------------
static inline size_t align256(size_t x) { return (x + 255) & ~(size_t)255; }

extern "C" void kernel_launch(void* const* d_in, const int* in_sizes, int n_in,
                              void* d_out, int out_size, void* d_ws, size_t ws_size,
                              hipStream_t stream)
{
    const float* test  = (const float*)d_in[0];   // [B, 512]
    const float* train = (const float*)d_in[1];   // [N, 512]
    float* out = (float*)d_out;                   // [B]

    const int B = in_sizes[0] / KDIM;             // 1024
    const int N = in_sizes[1] / KDIM;             // 50000
    const int nChunks = (N + BN - 1) / BN;        // 391

    char* ws = (char*)d_ws;
    size_t offBsq   = 0;
    size_t offPart  = align256((size_t)N * sizeof(float));
    size_t offTestH = align256(offPart + (size_t)B * nChunks * sizeof(float2));
    size_t szTest   = align256((size_t)B * KDIM * sizeof(__bf16));
    size_t szTrain  = align256((size_t)N * KDIM * sizeof(__bf16));
    size_t offTestL  = offTestH + szTest;
    size_t offTrainH = offTestL + szTest;
    size_t offTrainL = offTrainH + szTrain;
    size_t need      = offTrainL + szTrain;

    float*  bsq     = (float*)(ws + offBsq);
    float2* partial = (float2*)(ws + offPart);

    dim3 grid(nChunks, (B + BM - 1) / BM);

    if (ws_size >= need) {
        __bf16* testHi  = (__bf16*)(ws + offTestH);
        __bf16* testLo  = (__bf16*)(ws + offTestL);
        __bf16* trainHi = (__bf16*)(ws + offTrainH);
        __bf16* trainLo = (__bf16*)(ws + offTrainL);

        split_rows<<<(N + WAVES - 1) / WAVES, 256, 0, stream>>>(
            train, trainHi, trainLo, bsq, N);                 // fused ||x_n||^2
        split_rows<<<(B + WAVES - 1) / WAVES, 256, 0, stream>>>(
            test, testHi, testLo, (float*)nullptr, B);

        knn_cross_top2<true><<<grid, 256, 0, stream>>>(
            nullptr, nullptr, testHi, testLo, trainHi, trainLo,
            bsq, partial, B, N, nChunks);
    } else {
        row_sumsq<<<(N + WAVES - 1) / WAVES, 256, 0, stream>>>(train, bsq, N);
        knn_cross_top2<false><<<grid, 256, 0, stream>>>(
            test, train, nullptr, nullptr, nullptr, nullptr,
            bsq, partial, B, N, nChunks);
    }

    knn_finalize<<<(B + WAVES - 1) / WAVES, 256, 0, stream>>>(test, partial, out, B, nChunks);
}
